// HOTFormerBlock_70265664962587
// MI455X (gfx1250) — matrix operand order repaired and checked by reading, not compiled, not run on hardware
//
#include <hip/hip_runtime.h>
#include <hip/hip_bf16.h>
#include <math.h>

typedef __bf16 bf16_t;
typedef __attribute__((ext_vector_type(16))) __bf16 v16bf;
typedef __attribute__((ext_vector_type(8)))  float  v8f;

#define NPTS   131072      // N
#define WWIN   4096
#define CCH    256
#define KCN    27
#define SLEN   33
#define RTOK   135168      // WWIN * SLEN

// CDNA5 async global->LDS copy path (guarded; falls back to sync staging)
#if defined(__gfx1250__) && __has_builtin(__builtin_amdgcn_global_load_async_to_lds_b128) && __has_builtin(__builtin_amdgcn_s_wait_asynccnt)
#define HAVE_ASYNC 1
typedef int v4i_vec __attribute__((vector_size(16)));
typedef v4i_vec __attribute__((address_space(1)))* gv4i_p;   // global int4*
typedef v4i_vec __attribute__((address_space(3)))* lv4i_p;   // LDS int4*
#else
#define HAVE_ASYNC 0
#endif

__device__ __forceinline__ float gelu_tanh(float x) {
    float x3 = x * x * x;
    return 0.5f * x * (1.0f + tanhf(0.7978845608028654f * (x + 0.044715f * x3)));
}

__device__ __forceinline__ v16bf make_v16(uint4 lo, uint4 hi) {
    union { uint4 u[2]; v16bf v; } t;
    t.u[0] = lo; t.u[1] = hi;
    return t.v;
}

// ---------------------------------------------------------------- fp32 -> bf16
__global__ void cvt_bf16_kernel(const float* __restrict__ in, bf16_t* __restrict__ out, int n) {
    int i = blockIdx.x * blockDim.x + threadIdx.x;
    if (i < n) out[i] = (bf16_t)in[i];
}

// ------------------------------------- fp32 W[K][N] -> bf16 W^T[N][K] (one-time)
__global__ void cvtT_kernel(const float* __restrict__ in, bf16_t* __restrict__ outT,
                            int K, int N) {
    int i = blockIdx.x * blockDim.x + threadIdx.x;
    if (i < K * N) {
        int n = i / K, k = i - n * K;
        outT[i] = (bf16_t)in[(size_t)k * N + n];
    }
}

// ------------------------------------------------- CPE gather + BN + residual
__global__ void __launch_bounds__(256) cpe_kernel(
        const float* __restrict__ data, const int* __restrict__ neigh,
        const float* __restrict__ cpe_w,
        const float* __restrict__ gamma, const float* __restrict__ beta,
        const float* __restrict__ mean,  const float* __restrict__ var,
        float* __restrict__ X) {
    __shared__ __align__(16) float ldsW[KCN * CCH];
    __shared__ int ldsN[8 * KCN];
    int tid = threadIdx.x;
    for (int k = 0; k < KCN; ++k) ldsW[k * CCH + tid] = cpe_w[k * CCH + tid];
    int row0 = blockIdx.x * 8;
    if (tid < 8 * KCN) ldsN[tid] = neigh[(size_t)row0 * KCN + tid];
    __syncthreads();

    int rl = tid >> 5, lane = tid & 31;
    int n = row0 + rl;
    int cb = lane * 8;
    float acc[8];
#pragma unroll
    for (int j = 0; j < 8; ++j) acc[j] = 0.0f;
    for (int k = 0; k < KCN; ++k) {
        int idx = ldsN[rl * KCN + k];
        const float4* dp = (const float4*)(data + (size_t)idx * CCH + cb);
        const float4* wp = (const float4*)(ldsW + k * CCH + cb);
        float4 d0 = dp[0], d1 = dp[1];
        float4 w0 = wp[0], w1 = wp[1];
        acc[0] += d0.x * w0.x; acc[1] += d0.y * w0.y;
        acc[2] += d0.z * w0.z; acc[3] += d0.w * w0.w;
        acc[4] += d1.x * w1.x; acc[5] += d1.y * w1.y;
        acc[6] += d1.z * w1.z; acc[7] += d1.w * w1.w;
    }
    int w = n >> 5, s = n & 31;
    float* xp = X + ((size_t)w * SLEN + 1 + s) * CCH + cb;
    const float* dn = data + (size_t)n * CCH + cb;
    float o[8];
#pragma unroll
    for (int j = 0; j < 8; ++j) {
        int c = cb + j;
        float bn = (acc[j] - mean[c]) * rsqrtf(var[c] + 1e-5f) * gamma[c] + beta[c];
        o[j] = dn[j] + bn;
    }
    ((float4*)xp)[0] = make_float4(o[0], o[1], o[2], o[3]);
    ((float4*)xp)[1] = make_float4(o[4], o[5], o[6], o[7]);
}

__global__ void relay_kernel(const float* __restrict__ relay, float* __restrict__ X) {
    int w = blockIdx.x, c = threadIdx.x;
    X[(size_t)w * SLEN * CCH + c] = relay[(size_t)w * CCH + c];
}

// --------------------------------------------------------- LayerNorm -> bf16
__global__ void __launch_bounds__(256) ln_kernel(
        const float* __restrict__ in, const float* __restrict__ g,
        const float* __restrict__ b, bf16_t* __restrict__ out) {
    int wave = threadIdx.x >> 5, lane = threadIdx.x & 31;
    int row = blockIdx.x * 8 + wave;
    const float4* xv = (const float4*)(in + (size_t)row * CCH);
    float4 a0 = xv[lane * 2], a1 = xv[lane * 2 + 1];
    float v[8] = {a0.x, a0.y, a0.z, a0.w, a1.x, a1.y, a1.z, a1.w};
    float s = 0.f, s2 = 0.f;
#pragma unroll
    for (int j = 0; j < 8; ++j) { s += v[j]; s2 += v[j] * v[j]; }
#pragma unroll
    for (int m = 1; m < 32; m <<= 1) {
        s  += __shfl_xor(s,  m);
        s2 += __shfl_xor(s2, m);
    }
    float mu  = s * (1.0f / 256.0f);
    float var = s2 * (1.0f / 256.0f) - mu * mu;
    float inv = rsqrtf(var + 1e-5f);
    float4 g0 = ((const float4*)g)[lane * 2], g1 = ((const float4*)g)[lane * 2 + 1];
    float4 b0 = ((const float4*)b)[lane * 2], b1 = ((const float4*)b)[lane * 2 + 1];
    float gg[8] = {g0.x, g0.y, g0.z, g0.w, g1.x, g1.y, g1.z, g1.w};
    float bb[8] = {b0.x, b0.y, b0.z, b0.w, b1.x, b1.y, b1.z, b1.w};
    uint4 u;
    bf16_t* tp = (bf16_t*)&u;
#pragma unroll
    for (int j = 0; j < 8; ++j)
        tp[j] = (bf16_t)((v[j] - mu) * inv * gg[j] + bb[j]);
    ((uint4*)(out + (size_t)row * CCH))[lane] = u;
}

// ------------------------------------------------------------ WMMA bf16 GEMM
// Per-wave compute on one staged tile pair: A[128x32] row-major, Bt[128x32]=[n][k]
__device__ __forceinline__ void gemm_compute(
        const bf16_t* lA, const bf16_t* lBt, v8f (&acc)[8],
        int wave, int m, int hl) {
    const bf16_t* ap = lA + (wave * 16 + m) * 32;
    v16bf af = make_v16(*(const uint4*)(ap + hl * 8),
                        *(const uint4*)(ap + 16 + hl * 8));
#pragma unroll
    for (int nt = 0; nt < 8; ++nt) {
        v16bf bfr = make_v16(*(const uint4*)(lBt + (nt * 16 + m) * 32 + hl * 8),
                             *(const uint4*)(lBt + (nt * 16 + m) * 32 + 16 + hl * 8));
        acc[nt] = __builtin_amdgcn_wmma_f32_16x16x32_bf16(
            false, af, false, bfr, (short)0, acc[nt], false, false);
    }
}

// C[M,N] = epilogue(A[M,K] @ W^T[N,K]^T + bias).  Block tile 128x128, 8 waves,
// wave owns 16 rows x 128 cols (8 WMMA accumulators), K-step 32, double-buffered
// async global->LDS staging when available.
template<int ACT, int RES, int OUTB, int K, int NOUT>
__global__ void __launch_bounds__(256) gemm_kernel(
        const bf16_t* __restrict__ A, const bf16_t* __restrict__ WT,
        const float* __restrict__ bias, const float* __restrict__ res,
        bf16_t* __restrict__ outb, float* __restrict__ outf) {
    __shared__ __align__(64) bf16_t ldsA[2][128 * 32];   // [buf][row][k]
    __shared__ __align__(64) bf16_t ldsBt[2][128 * 32];  // [buf][n][k]
    int tid  = threadIdx.x;
    int m0   = blockIdx.x * 128;
    int n0   = blockIdx.y * 128;
    int wave = tid >> 5, lane = tid & 31;
    int m = lane & 15, hl = lane >> 4;

    v8f zero8 = {0.f, 0.f, 0.f, 0.f, 0.f, 0.f, 0.f, 0.f};
    v8f acc[8];
#pragma unroll
    for (int i = 0; i < 8; ++i) acc[i] = zero8;

    int srow = tid >> 1, shalf = tid & 1;   // staging: 2 threads/row, 16 bf16 each
    const bf16_t* Ab = A  + (size_t)(m0 + srow) * K + shalf * 16;
    const bf16_t* Bb = WT + (size_t)(n0 + srow) * K + shalf * 16;
    int loff = srow * 32 + shalf * 16;

#if HAVE_ASYNC
    // -------- double-buffered async staging --------
    auto issue = [&](int k0, int buf) {
        __builtin_amdgcn_global_load_async_to_lds_b128(
            (gv4i_p)(Ab + k0),     (lv4i_p)&ldsA[buf][loff],      0, 0);
        __builtin_amdgcn_global_load_async_to_lds_b128(
            (gv4i_p)(Ab + k0 + 8), (lv4i_p)&ldsA[buf][loff + 8],  0, 0);
        __builtin_amdgcn_global_load_async_to_lds_b128(
            (gv4i_p)(Bb + k0),     (lv4i_p)&ldsBt[buf][loff],     0, 0);
        __builtin_amdgcn_global_load_async_to_lds_b128(
            (gv4i_p)(Bb + k0 + 8), (lv4i_p)&ldsBt[buf][loff + 8], 0, 0);
    };
    issue(0, 0);
    __builtin_amdgcn_s_wait_asynccnt(0);
    __syncthreads();
    int buf = 0;
    for (int k0 = 0; k0 < K; k0 += 32) {
        if (k0 + 32 < K) issue(k0 + 32, buf ^ 1);
        gemm_compute(ldsA[buf], ldsBt[buf], acc, wave, m, hl);
        if (k0 + 32 < K) __builtin_amdgcn_s_wait_asynccnt(0);
        __syncthreads();
        buf ^= 1;
    }
#else
    // -------- sync staging fallback --------
    for (int k0 = 0; k0 < K; k0 += 32) {
        {
            const uint4* sa = (const uint4*)(Ab + k0);
            uint4* da = (uint4*)(ldsA[0] + loff);
            da[0] = sa[0]; da[1] = sa[1];
            const uint4* sb = (const uint4*)(Bb + k0);
            uint4* db = (uint4*)(ldsBt[0] + loff);
            db[0] = sb[0]; db[1] = sb[1];
        }
        if (k0 + 32 < K) {
            __builtin_prefetch(Ab + k0 + 32, 0, 1);
            __builtin_prefetch(Bb + k0 + 32, 0, 1);
        }
        __syncthreads();
        gemm_compute(ldsA[0], ldsBt[0], acc, wave, m, hl);
        __syncthreads();
    }
#endif

    // epilogue: bias (+gelu) (+residual), fp32 or bf16 store
#pragma unroll
    for (int nt = 0; nt < 8; ++nt) {
#pragma unroll
        for (int r = 0; r < 8; ++r) {
            int row = m0 + wave * 16 + r + 8 * hl;
            int col = n0 + nt * 16 + m;
            float v = acc[nt][r] + bias[col];
            if (ACT == 1) v = gelu_tanh(v);
            if (RES)  v += res[(size_t)row * NOUT + col];
            if (OUTB) outb[(size_t)row * NOUT + col] = (bf16_t)v;
            else      outf[(size_t)row * NOUT + col] = v;
        }
    }
}

// -------------------------------------------------------- windowed attention
// one block per window; wave = head.  S=33 padded to 48 (3 tiles), D=32.
__global__ void __launch_bounds__(256) attn_kernel(
        const bf16_t* __restrict__ qkv, bf16_t* __restrict__ attn_out) {
    // pool: [ Vt 256x48 | (Qb 33x256 | Kb 33x256)  overlapped by  Pb 8x48x48 ]
    __shared__ __align__(64) char pool[24576 + 36864];
    bf16_t* Vt = (bf16_t*)pool;                  // [d_chan][t], stride 48
    bf16_t* Qb = (bf16_t*)(pool + 24576);        // [s][c]
    bf16_t* Kb = (bf16_t*)(pool + 24576 + 16896);
    bf16_t* Pb = (bf16_t*)(pool + 24576);        // [head][row 48][col 48]

    int w    = blockIdx.x;
    int tid  = threadIdx.x;
    int wave = tid >> 5;   // head
    int lane = tid & 31;
    int h    = wave;
    int m = lane & 15, hl = lane >> 4;
    const size_t base = (size_t)w * SLEN * 768;
    const float scale = 0.17677669529663687f; // 32^-0.5

    // ---- stage: de-interleave qkv window (33x768) into Qb / Kb / Vt ----
    for (int i = tid; i < (SLEN * 768) / 8; i += 256) {
        int s = i / 96;
        int c = (i % 96) * 8;
        uint4 v = *(const uint4*)(qkv + base + (size_t)s * 768 + c);
        if (c < 256) {
            *(uint4*)(Qb + s * 256 + c) = v;
        } else if (c < 512) {
            *(uint4*)(Kb + s * 256 + (c - 256)) = v;
        } else {
            int cl = c - 512;
            const bf16_t* tp = (const bf16_t*)&v;
#pragma unroll
            for (int ii = 0; ii < 8; ++ii)
                Vt[(cl + ii) * 48 + s] = tp[ii];
        }
    }
    for (int i = tid; i < 256 * 15; i += 256) {   // zero V-pad cols 33..47
        int d = i / 15, col = 33 + i % 15;
        Vt[d * 48 + col] = (bf16_t)0.f;
    }
    __syncthreads();

    // ---- logits = Q @ K^T ----
    v16bf aq[3];
#pragma unroll
    for (int mt = 0; mt < 3; ++mt) {
        int sc = min(mt * 16 + m, SLEN - 1);
        const bf16_t* qp = Qb + sc * 256 + h * 32;
        aq[mt] = make_v16(*(const uint4*)(qp + hl * 8),
                          *(const uint4*)(qp + 16 + hl * 8));
    }
    v8f zero8 = {0.f, 0.f, 0.f, 0.f, 0.f, 0.f, 0.f, 0.f};
    v8f L[3][3];
#pragma unroll
    for (int a = 0; a < 3; ++a)
#pragma unroll
        for (int b = 0; b < 3; ++b) L[a][b] = zero8;
#pragma unroll
    for (int nt = 0; nt < 3; ++nt) {
        int tc = min(nt * 16 + m, SLEN - 1);
        v16bf bk = make_v16(*(const uint4*)(Kb + tc * 256 + h * 32 + hl * 8),
                            *(const uint4*)(Kb + tc * 256 + h * 32 + 16 + hl * 8));
#pragma unroll
        for (int mt = 0; mt < 3; ++mt)
            L[mt][nt] = __builtin_amdgcn_wmma_f32_16x16x32_bf16(
                false, aq[mt], false, bk, (short)0, L[mt][nt], false, false);
    }
    __syncthreads();   // all waves done reading Qb/Kb before Pb overwrites them

    // ---- softmax (cols 0..32 valid; tile2 valid only at m==0) ----
#pragma unroll
    for (int mt = 0; mt < 3; ++mt) {
#pragma unroll
        for (int r = 0; r < 8; ++r) {
            float v0 = L[mt][0][r] * scale;
            float v1 = L[mt][1][r] * scale;
            float v2 = (m == 0) ? L[mt][2][r] * scale : -1e30f;
            float mx = fmaxf(fmaxf(v0, v1), v2);
#pragma unroll
            for (int msk = 1; msk < 16; msk <<= 1) mx = fmaxf(mx, __shfl_xor(mx, msk));
            float e0 = __expf(v0 - mx), e1 = __expf(v1 - mx), e2 = __expf(v2 - mx);
            float sum = e0 + e1 + e2;
#pragma unroll
            for (int msk = 1; msk < 16; msk <<= 1) sum += __shfl_xor(sum, msk);
            float inv = 1.0f / sum;
            int row = mt * 16 + r + 8 * hl;
            bf16_t* pr = Pb + h * 2304 + row * 48;
            pr[0 * 16 + m] = (bf16_t)(e0 * inv);
            pr[1 * 16 + m] = (bf16_t)(e1 * inv);
            pr[2 * 16 + m] = (bf16_t)(e2 * inv);    // ==0 for masked cols
        }
    }
    __syncthreads();

    // ---- out = P(48x64pad) @ V : 2 K-steps of 32 ----
    uint4 z16 = make_uint4(0, 0, 0, 0);
    v8f O[3][2];
#pragma unroll
    for (int a = 0; a < 3; ++a) { O[a][0] = zero8; O[a][1] = zero8; }
#pragma unroll
    for (int ks = 0; ks < 2; ++ks) {
        v16bf ap[3];
#pragma unroll
        for (int mt = 0; mt < 3; ++mt) {
            const bf16_t* pp = Pb + h * 2304 + (mt * 16 + m) * 48;
            if (ks == 0)
                ap[mt] = make_v16(*(const uint4*)(pp + hl * 8),
                                  *(const uint4*)(pp + 16 + hl * 8));
            else  // k=32..47 valid (j<4 slots), k=48..63 compile-time zero
                ap[mt] = make_v16(*(const uint4*)(pp + 32 + hl * 8), z16);
        }
        int cb = ks * 32 + hl * 8;   // V^T column base (finite pad)
#pragma unroll
        for (int nt = 0; nt < 2; ++nt) {
            v16bf bv = make_v16(*(const uint4*)(Vt + (h * 32 + nt * 16 + m) * 48 + cb),
                                *(const uint4*)(Vt + (h * 32 + nt * 16 + m) * 48 + min(cb + 16, 40)));
#pragma unroll
            for (int mt = 0; mt < 3; ++mt)
                O[mt][nt] = __builtin_amdgcn_wmma_f32_16x16x32_bf16(
                    false, ap[mt], false, bv, (short)0, O[mt][nt], false, false);
        }
    }

    // ---- store valid rows, channel layout = head-major (h*32 + d) ----
#pragma unroll
    for (int mt = 0; mt < 3; ++mt)
#pragma unroll
        for (int r = 0; r < 8; ++r) {
            int row = mt * 16 + r + 8 * hl;
            if (row < SLEN) {
                size_t o = ((size_t)w * SLEN + row) * CCH + h * 32;
                attn_out[o + 0 * 16 + m] = (bf16_t)O[mt][0][r];
                attn_out[o + 1 * 16 + m] = (bf16_t)O[mt][1][r];
            }
        }
}

// ----------------------------------------------------- split x3 into outputs
__global__ void out_kernel(const float* __restrict__ X3, float* __restrict__ out) {
    int row = blockIdx.x, c = threadIdx.x;
    int w = row / SLEN, s = row % SLEN;
    float v = X3[(size_t)row * CCH + c];
    if (s == 0) out[(size_t)NPTS * CCH + (size_t)w * CCH + c] = v;
    else        out[((size_t)w * 32 + (s - 1)) * CCH + c] = v;
}

// ============================================================================
extern "C" void kernel_launch(void* const* d_in, const int* in_sizes, int n_in,
                              void* d_out, int out_size, void* d_ws, size_t ws_size,
                              hipStream_t stream) {
    const float* data    = (const float*)d_in[0];
    const float* relay   = (const float*)d_in[1];
    const int*   neigh   = (const int*)  d_in[2];
    const float* cpe_w   = (const float*)d_in[3];
    const float* bn_g    = (const float*)d_in[4];
    const float* bn_b    = (const float*)d_in[5];
    const float* bn_m    = (const float*)d_in[6];
    const float* bn_v    = (const float*)d_in[7];
    const float* n1g     = (const float*)d_in[8];
    const float* n1b     = (const float*)d_in[9];
    const float* qkv_w   = (const float*)d_in[10];
    const float* qkv_b   = (const float*)d_in[11];
    const float* proj_w  = (const float*)d_in[12];
    const float* proj_b  = (const float*)d_in[13];
    const float* n2g     = (const float*)d_in[14];
    const float* n2b     = (const float*)d_in[15];
    const float* fc1_w   = (const float*)d_in[16];
    const float* fc1_b   = (const float*)d_in[17];
    const float* fc2_w   = (const float*)d_in[18];
    const float* fc2_b   = (const float*)d_in[19];
    float* out = (float*)d_out;

    // ---- workspace carve-up (weights stored transposed: W^T[N][K] bf16) ----
    char* p = (char*)d_ws;
    bf16_t* wqT = (bf16_t*)p; p += (size_t)768 * 256 * 2;
    bf16_t* wpT = (bf16_t*)p; p += (size_t)256 * 256 * 2;
    bf16_t* w1T = (bf16_t*)p; p += (size_t)1024 * 256 * 2;
    bf16_t* w2T = (bf16_t*)p; p += (size_t)256 * 1024 * 2;
    float*  X   = (float*) p; p += (size_t)RTOK * 256 * 4;   // x / x3 (aliased)
    float*  X2  = (float*) p; p += (size_t)RTOK * 256 * 4;
    bf16_t* Hb  = (bf16_t*)p; p += (size_t)RTOK * 256 * 2;   // h / attn_out / h2
    bf16_t* QKV = (bf16_t*)p; p += (size_t)RTOK * 768 * 2;
    bf16_t* F1  = (bf16_t*)p; p += (size_t)RTOK * 1024 * 2;
    (void)ws_size; (void)in_sizes; (void)n_in; (void)out_size;

    // 0) weights -> bf16, transposed to [N][K]
    cvtT_kernel<<<(256 * 768 + 255) / 256,  256, 0, stream>>>(qkv_w,  wqT, 256, 768);
    cvtT_kernel<<<(256 * 256 + 255) / 256,  256, 0, stream>>>(proj_w, wpT, 256, 256);
    cvtT_kernel<<<(256 * 1024 + 255) / 256, 256, 0, stream>>>(fc1_w,  w1T, 256, 1024);
    cvtT_kernel<<<(1024 * 256 + 255) / 256, 256, 0, stream>>>(fc2_w,  w2T, 1024, 256);

    // 1) CPE + BN + residual -> X[:,1:,:] ; relay tokens -> X[:,0,:]
    cpe_kernel<<<NPTS / 8, 256, 0, stream>>>(data, neigh, cpe_w, bn_g, bn_b, bn_m, bn_v, X);
    relay_kernel<<<WWIN, 256, 0, stream>>>(relay, X);

    // 2) LN1 -> Hb (bf16)
    ln_kernel<<<RTOK / 8, 256, 0, stream>>>(X, n1g, n1b, Hb);

    // 3) QKV GEMM
    gemm_kernel<0, 0, 1, 256, 768><<<dim3(RTOK / 128, 768 / 128), 256, 0, stream>>>(
        Hb, wqT, qkv_b, nullptr, QKV, nullptr);

    // 4) windowed attention -> Hb (reused as attn_out)
    attn_kernel<<<WWIN, 256, 0, stream>>>(QKV, Hb);

    // 5) proj GEMM + residual(X) -> X2 (fp32)
    gemm_kernel<0, 1, 0, 256, 256><<<dim3(RTOK / 128, 256 / 128), 256, 0, stream>>>(
        Hb, wpT, proj_b, X, nullptr, X2);

    // 6) LN2 -> Hb (reused as h2)
    ln_kernel<<<RTOK / 8, 256, 0, stream>>>(X2, n2g, n2b, Hb);

    // 7) FC1 GEMM + GELU -> F1 (bf16)
    gemm_kernel<1, 0, 1, 256, 1024><<<dim3(RTOK / 128, 1024 / 128), 256, 0, stream>>>(
        Hb, w1T, fc1_b, nullptr, F1, nullptr);

    // 8) FC2 GEMM + residual(X2) -> X (reused as x3)
    gemm_kernel<0, 1, 0, 1024, 256><<<dim3(RTOK / 128, 256 / 128), 256, 0, stream>>>(
        F1, w2T, fc2_b, X2, nullptr, X);

    // 9) split into (data_out, rt)
    out_kernel<<<RTOK, 256, 0, stream>>>(X, out);
}
